// SwinBlock_23630910063016
// MI455X (gfx1250) — matrix operand order repaired and checked by reading, not compiled, not run on hardware
//
#include <hip/hip_runtime.h>
#include <cstdint>

typedef unsigned short u16;
typedef __attribute__((ext_vector_type(16))) __bf16 v16bf;
typedef __attribute__((ext_vector_type(8)))  float  v8f;

union FragBf { uint4 q[2]; v16bf v; };

__device__ __forceinline__ u16 f2bf(float f) {
    union { float f; unsigned u; } x; x.f = f;
    unsigned r = x.u + 0x7fffu + ((x.u >> 16) & 1u);
    return (u16)(r >> 16);
}

// --- CDNA5 async global->LDS path (ASYNCcnt), guarded for toolchain drift ---
#if __has_builtin(__builtin_amdgcn_global_load_async_to_lds_b128)
#define HAVE_ASYNC_LDS 1
#else
#define HAVE_ASYNC_LDS 0
#endif

// Builtin signature (probe-confirmed by round-2 diagnostic): parameters are
// pointers to a 16-byte int vector in AS1 (global) / AS3 (LDS).
typedef int v4i_t __attribute__((vector_size(16)));
typedef __attribute__((address_space(1))) v4i_t as1_v4i;
typedef __attribute__((address_space(3))) v4i_t as3_v4i;

#if HAVE_ASYNC_LDS
#if __has_builtin(__builtin_amdgcn_s_wait_asynccnt)
#define WAIT_ASYNC(n) __builtin_amdgcn_s_wait_asynccnt(n)
#else
#define WAIT_ASYNC(n) asm volatile("s_wait_asynccnt %0" ::"i"(n) : "memory")
#endif
#else
#define WAIT_ASYNC(n) ((void)0)
#endif

// ---------------------------------------------------------------------------
// Weight convert + transpose: in f32 [K][N] -> out bf16 [N][K]
// ---------------------------------------------------------------------------
__global__ void cvt_transpose_kernel(const float* __restrict__ in, u16* __restrict__ out,
                                     int K, int N) {
    int t = blockIdx.x * 256 + threadIdx.x;
    if (t >= K * N) return;
    int k = t / N, n = t - k * N;
    out[(size_t)n * K + k] = f2bf(in[t]);
}

// ---------------------------------------------------------------------------
// LayerNorm f32 [M][384] -> bf16 [M][384]. One wave (32 lanes) per row.
// ---------------------------------------------------------------------------
__global__ void ln_bf16_kernel(const float* __restrict__ x, const float* __restrict__ g,
                               const float* __restrict__ b, u16* __restrict__ out, int M) {
    int wid = threadIdx.x >> 5, lane = threadIdx.x & 31;
    int row = blockIdx.x * 8 + wid;
    if (row >= M) return;
    const float* xr = x + (size_t)row * 384;
    float vals[12], s = 0.f, sq = 0.f;
#pragma unroll
    for (int j = 0; j < 12; j++) { float v = xr[lane + 32 * j]; vals[j] = v; s += v; sq += v * v; }
#pragma unroll
    for (int m = 16; m >= 1; m >>= 1) { s += __shfl_xor(s, m); sq += __shfl_xor(sq, m); }
    float mu = s * (1.f / 384.f);
    float var = sq * (1.f / 384.f) - mu * mu;
    float rstd = rsqrtf(var + 1e-5f);
    u16* orow = out + (size_t)row * 384;
#pragma unroll
    for (int j = 0; j < 12; j++) {
        int c = lane + 32 * j;
        orow[c] = f2bf((vals[j] - mu) * rstd * g[c] + b[c]);
    }
}

// ---------------------------------------------------------------------------
// bf16 GEMM: A [M][K] row-major bf16, Bt [N][K] bf16 (pre-transposed weights).
// Block = 256 thr = 8 waves, tile 128x128, BK=32. Wave -> 32x64 (2x4 WMMA tiles).
// Double-buffered LDS; tiles prefetched with GLOBAL_LOAD_ASYNC_TO_LDS_B128
// (4 per thread per tile; async loads retire in order, so s_wait_asynccnt<=4
// guarantees the current tile while the next stays in flight).
// MODE 0: bias -> bf16 out;  MODE 1: bias + residual -> f32 out;
// MODE 2: bias + exact GELU -> bf16 out.
// ---------------------------------------------------------------------------
template <int MODE>
__global__ __launch_bounds__(256)
void gemm_bf16_kernel(const u16* __restrict__ A, const u16* __restrict__ Bt,
                      const float* __restrict__ bias, const float* __restrict__ res,
                      void* __restrict__ outv, int M, int N, int K) {
    // 80-byte padded rows (40 bf16): 80/4=20 words, gcd(20,64)=4 -> rows 0..15 hit
    // distinct bank groups for the b128 fragment gathers.
    __shared__ __align__(16) u16 As[2][128 * 40];
    __shared__ __align__(16) u16 Bs[2][128 * 40];

    const int m0 = blockIdx.y * 128;
    const int n0 = blockIdx.x * 128;
    const int tid = threadIdx.x;
    const int wid = tid >> 5, lane = tid & 31;
    const int wm = wid >> 1, wn = wid & 1;
    const int l16 = lane & 15, hi = lane >> 4;
    const int lrow = tid >> 1, lhalf = tid & 1;

    // Per-thread slice: 16 bf16 (32B) of one A row and one B row per K-tile.
    const u16* agp = A + (size_t)(m0 + lrow) * K + lhalf * 16;
    const u16* bgp = Bt + (size_t)(n0 + lrow) * K + lhalf * 16;
    const int lds_off = lrow * 40 + lhalf * 16;  // elements

    auto issue_tile = [&](int kk, int buf) {
#if HAVE_ASYNC_LDS
        __builtin_amdgcn_global_load_async_to_lds_b128(
            (as1_v4i*)(agp + kk), (as3_v4i*)&As[buf][lds_off], 0, 0);
        __builtin_amdgcn_global_load_async_to_lds_b128(
            (as1_v4i*)(agp + kk + 8), (as3_v4i*)&As[buf][lds_off + 8], 0, 0);
        __builtin_amdgcn_global_load_async_to_lds_b128(
            (as1_v4i*)(bgp + kk), (as3_v4i*)&Bs[buf][lds_off], 0, 0);
        __builtin_amdgcn_global_load_async_to_lds_b128(
            (as1_v4i*)(bgp + kk + 8), (as3_v4i*)&Bs[buf][lds_off + 8], 0, 0);
#else
        const uint4* ap = (const uint4*)(agp + kk);
        uint4 a0 = ap[0], a1 = ap[1];
        const uint4* bp = (const uint4*)(bgp + kk);
        uint4 b0 = bp[0], b1 = bp[1];
        uint4* ad = (uint4*)&As[buf][lds_off];
        ad[0] = a0; ad[1] = a1;
        uint4* bd = (uint4*)&Bs[buf][lds_off];
        bd[0] = b0; bd[1] = b1;
#endif
    };

    v8f acc[2][4];
#pragma unroll
    for (int mt = 0; mt < 2; mt++)
#pragma unroll
        for (int nt = 0; nt < 4; nt++)
#pragma unroll
            for (int i = 0; i < 8; i++) acc[mt][nt][i] = 0.f;

    const int nk = K >> 5;
    issue_tile(0, 0);

    for (int kt = 0; kt < nk; kt++) {
        const int cur = kt & 1;
        const bool pf = (kt + 1 < nk);
        if (pf) issue_tile((kt + 1) << 5, cur ^ 1);
        if (pf) { WAIT_ASYNC(4); } else { WAIT_ASYNC(0); }
        __syncthreads();  // tile kt resident for all waves

        FragBf fa[2], fb[4];
#pragma unroll
        for (int mt = 0; mt < 2; mt++) {
            // 16-bit A layout (ISA 7.12.2): lane holds K = {hi*8..+7, 16+hi*8..+7}
            const char* base = (const char*)&As[cur][0] + (wm * 32 + mt * 16 + l16) * 80 + hi * 16;
            fa[mt].q[0] = *(const uint4*)base;
            fa[mt].q[1] = *(const uint4*)(base + 32);
        }
#pragma unroll
        for (int nt = 0; nt < 4; nt++) {
            // B layout: lane = column n, holds K = {hi*8..+7, 16+hi*8..+7}
            const char* base = (const char*)&Bs[cur][0] + (wn * 64 + nt * 16 + l16) * 80 + hi * 16;
            fb[nt].q[0] = *(const uint4*)base;
            fb[nt].q[1] = *(const uint4*)(base + 32);
        }
#pragma unroll
        for (int mt = 0; mt < 2; mt++)
#pragma unroll
            for (int nt = 0; nt < 4; nt++)
                acc[mt][nt] = __builtin_amdgcn_wmma_f32_16x16x32_bf16(
                    false, fa[mt].v, false, fb[nt].v, (short)0, acc[mt][nt], false, false);

        __syncthreads();  // all waves done reading buf[cur]; safe to overwrite next iter
    }

    // Epilogue. C layout: row = 8*hi + v, col = l16 (within tile).
#pragma unroll
    for (int mt = 0; mt < 2; mt++) {
        int rb = m0 + wm * 32 + mt * 16 + hi * 8;
#pragma unroll
        for (int nt = 0; nt < 4; nt++) {
            int col = n0 + wn * 64 + nt * 16 + l16;
            float bv = bias[col];
#pragma unroll
            for (int v = 0; v < 8; v++) {
                size_t idx = (size_t)(rb + v) * N + col;
                float x = acc[mt][nt][v] + bv;
                if (MODE == 0) {
                    ((u16*)outv)[idx] = f2bf(x);
                } else if (MODE == 1) {
                    ((float*)outv)[idx] = x + res[idx];
                } else {
                    float ge = 0.5f * x * (1.f + erff(x * 0.70710678118654752f));
                    ((u16*)outv)[idx] = f2bf(ge);
                }
            }
        }
    }
}

// ---------------------------------------------------------------------------
// Windowed attention. Block = 128 thr (4 waves) per (window, head).
// 49 tokens padded to 64. Shift (roll -3) folded into the gather index.
// qkv: bf16 [token][1152] ({q,k,v} x 12 heads x 32). out: bf16 [token][384].
// ---------------------------------------------------------------------------
__global__ __launch_bounds__(128)
void attn_kernel(const u16* __restrict__ qkv, u16* __restrict__ out) {
    __shared__ __align__(16) u16 qs[64 * 40];     // [query][dim], 80B rows
    __shared__ __align__(16) u16 ks[64 * 40];     // [key][dim]
    __shared__ __align__(16) u16 vst[32 * 72];    // V^T: [dim][key], 144B rows
    __shared__ __align__(16) u16 ps[4 * 16 * 72]; // per-wave P: [16][64], 144B rows

    const int head = blockIdx.x % 12;
    const int win  = blockIdx.x / 12;
    const int batch = win >> 6;
    const int wr = (win >> 3) & 7, wc = win & 7;
    const int tid = threadIdx.x;

    {   // cooperative gather: thread = (token e, half of 32 dims)
        int e = tid >> 1, half = tid & 1;
        uint4 z; z.x = z.y = z.z = z.w = 0u;
        uint4 a0 = z, a1 = z, b0 = z, b1 = z, c0 = z, c1 = z;
        if (e < 49) {
            int i7 = e / 7, j7 = e - i7 * 7;
            int oh = (wr * 7 + i7 + 3) % 56;   // inverse of roll(-3)
            int ow = (wc * 7 + j7 + 3) % 56;
            size_t tok = (size_t)batch * 3136 + oh * 56 + ow;
            const u16* base = qkv + tok * 1152 + head * 32 + half * 16;
            const uint4* qp = (const uint4*)(base);
            const uint4* kp = (const uint4*)(base + 384);
            const uint4* vp = (const uint4*)(base + 768);
            a0 = qp[0]; a1 = qp[1];
            b0 = kp[0]; b1 = kp[1];
            c0 = vp[0]; c1 = vp[1];
        }
        uint4* qd = (uint4*)((char*)qs + e * 80 + half * 32); qd[0] = a0; qd[1] = a1;
        uint4* kd = (uint4*)((char*)ks + e * 80 + half * 32); kd[0] = b0; kd[1] = b1;
        union { uint4 q[2]; u16 u[16]; } vv; vv.q[0] = c0; vv.q[1] = c1;
#pragma unroll
        for (int d2 = 0; d2 < 16; d2++) vst[(half * 16 + d2) * 72 + e] = vv.u[d2];
    }
    __syncthreads();

    const int w = tid >> 5, lane = tid & 31;
    const int l16 = lane & 15, hi = lane >> 4;

    // S = Q(16x32) @ K^T(32x64): one A frag, four B frags.
    FragBf fq;
    {
        const char* base = (const char*)qs + (w * 16 + l16) * 80 + hi * 16;
        fq.q[0] = *(const uint4*)base;
        fq.q[1] = *(const uint4*)(base + 32);
    }
    v8f s[4];
#pragma unroll
    for (int nt = 0; nt < 4; nt++) {
        FragBf fk;
        const char* base = (const char*)ks + (nt * 16 + l16) * 80 + hi * 16;
        fk.q[0] = *(const uint4*)base;
        fk.q[1] = *(const uint4*)(base + 32);
        v8f z; for (int i = 0; i < 8; i++) z[i] = 0.f;
        s[nt] = __builtin_amdgcn_wmma_f32_16x16x32_bf16(
            false, fq.v, false, fk.v, (short)0, z, false, false);
    }

    // Masked softmax. Row r = 8*hi+v lives in 16 lanes sharing hi -> xor masks 1..8.
    const float scale = 0.17677669529663687f; // 1/sqrt(32)
    float p[4][8];
#pragma unroll
    for (int v = 0; v < 8; v++) {
        float sv[4], mx = -1e30f;
#pragma unroll
        for (int nt = 0; nt < 4; nt++) {
            int col = nt * 16 + l16;
            float x = (col < 49) ? s[nt][v] * scale : -1e30f;
            sv[nt] = x;
            mx = fmaxf(mx, x);
        }
#pragma unroll
        for (int m = 8; m >= 1; m >>= 1) mx = fmaxf(mx, __shfl_xor(mx, m));
        float sum = 0.f;
#pragma unroll
        for (int nt = 0; nt < 4; nt++) { float e = __expf(sv[nt] - mx); p[nt][v] = e; sum += e; }
#pragma unroll
        for (int m = 8; m >= 1; m >>= 1) sum += __shfl_xor(sum, m);
        float inv = 1.f / sum;
#pragma unroll
        for (int nt = 0; nt < 4; nt++) p[nt][v] *= inv;
    }

    // C-layout -> A-fragment layout via per-wave LDS (same-wave DS ops are in order).
#pragma unroll
    for (int nt = 0; nt < 4; nt++)
#pragma unroll
        for (int v = 0; v < 8; v++)
            ps[(w * 16 + hi * 8 + v) * 72 + nt * 16 + l16] = f2bf(p[nt][v]);

    // O(16x32) = P(16x64) @ V(64x32): K split into two 32-chunks.
    v8f o[2];
    for (int i = 0; i < 8; i++) { o[0][i] = 0.f; o[1][i] = 0.f; }
#pragma unroll
    for (int kc = 0; kc < 2; kc++) {
        FragBf fp;
        const char* abase = (const char*)ps + (w * 16 + l16) * 144 + kc * 64 + hi * 16;
        fp.q[0] = *(const uint4*)abase;
        fp.q[1] = *(const uint4*)(abase + 32);
#pragma unroll
        for (int nt2 = 0; nt2 < 2; nt2++) {
            FragBf fv;
            const char* bbase = (const char*)vst + (nt2 * 16 + l16) * 144 + kc * 64 + hi * 16;
            fv.q[0] = *(const uint4*)bbase;
            fv.q[1] = *(const uint4*)(bbase + 32);
            o[nt2] = __builtin_amdgcn_wmma_f32_16x16x32_bf16(
                false, fp.v, false, fv.v, (short)0, o[nt2], false, false);
        }
    }

    // Scatter back to token order (un-rolls the shift).
#pragma unroll
    for (int v = 0; v < 8; v++) {
        int qrow = w * 16 + hi * 8 + v;
        if (qrow < 49) {
            int i7 = qrow / 7, j7 = qrow - i7 * 7;
            int oh = (wr * 7 + i7 + 3) % 56;
            int ow = (wc * 7 + j7 + 3) % 56;
            size_t tok = (size_t)batch * 3136 + oh * 56 + ow;
#pragma unroll
            for (int nt2 = 0; nt2 < 2; nt2++)
                out[tok * 384 + head * 32 + nt2 * 16 + l16] = f2bf(o[nt2][v]);
        }
    }
}

// ---------------------------------------------------------------------------
// Host-side orchestration
// ---------------------------------------------------------------------------
extern "C" void kernel_launch(void* const* d_in, const int* in_sizes, int n_in,
                              void* d_out, int out_size, void* d_ws, size_t ws_size,
                              hipStream_t stream) {
    const int M = 32 * 56 * 56;   // 100352 tokens
    const int C = 384, H3 = 1152, HID = 1536;

    const float* x     = (const float*)d_in[0];
    const float* g1    = (const float*)d_in[3];
    const float* b1    = (const float*)d_in[4];
    const float* wqkv  = (const float*)d_in[5];
    const float* bqkv  = (const float*)d_in[6];
    const float* wproj = (const float*)d_in[7];
    const float* bproj = (const float*)d_in[8];
    const float* g2    = (const float*)d_in[9];
    const float* b2    = (const float*)d_in[10];
    const float* wfc1  = (const float*)d_in[11];
    const float* bfc1  = (const float*)d_in[12];
    const float* wfc2  = (const float*)d_in[13];
    const float* bfc2  = (const float*)d_in[14];
    float* out = (float*)d_out;

    char* base = (char*)d_ws;
    size_t off = 0;
    auto alloc = [&](size_t bytes) -> char* {
        char* p = base + off;
        off = (off + bytes + 255) & ~(size_t)255;
        return p;
    };
    u16*   xln    = (u16*)alloc((size_t)M * C * 2);
    u16*   qkvb   = (u16*)alloc((size_t)M * H3 * 2);
    u16*   attnb  = (u16*)alloc((size_t)M * C * 2);
    float* xo     = (float*)alloc((size_t)M * C * 4);
    u16*   ln2b   = (u16*)alloc((size_t)M * C * 2);
    u16*   h1     = (u16*)alloc((size_t)M * HID * 2);
    u16*   wqkvT  = (u16*)alloc((size_t)H3 * C * 2);
    u16*   wprojT = (u16*)alloc((size_t)C * C * 2);
    u16*   wfc1T  = (u16*)alloc((size_t)HID * C * 2);
    u16*   wfc2T  = (u16*)alloc((size_t)C * HID * 2);
    (void)ws_size; (void)in_sizes; (void)n_in; (void)out_size;

    // Weight conversion (tiny, amortized over 100K-row GEMMs)
    cvt_transpose_kernel<<<(C * H3 + 255) / 256, 256, 0, stream>>>(wqkv, wqkvT, C, H3);
    cvt_transpose_kernel<<<(C * C + 255) / 256, 256, 0, stream>>>(wproj, wprojT, C, C);
    cvt_transpose_kernel<<<(C * HID + 255) / 256, 256, 0, stream>>>(wfc1, wfc1T, C, HID);
    cvt_transpose_kernel<<<(HID * C + 255) / 256, 256, 0, stream>>>(wfc2, wfc2T, HID, C);

    // LN1 -> bf16
    ln_bf16_kernel<<<M / 8, 256, 0, stream>>>(x, g1, b1, xln, M);

    // QKV: [M,384] x [384,1152] + bias -> bf16
    gemm_bf16_kernel<0><<<dim3(H3 / 128, M / 128), 256, 0, stream>>>(
        xln, wqkvT, bqkv, nullptr, qkvb, M, H3, C);

    // Windowed attention: 2048 windows x 12 heads
    attn_kernel<<<2048 * 12, 128, 0, stream>>>(qkvb, attnb);

    // Proj + residual(x) -> xo (f32)
    gemm_bf16_kernel<1><<<dim3(C / 128, M / 128), 256, 0, stream>>>(
        attnb, wprojT, bproj, x, xo, M, C, C);

    // LN2 -> bf16
    ln_bf16_kernel<<<M / 8, 256, 0, stream>>>(xo, g2, b2, ln2b, M);

    // FC1 + GELU -> bf16
    gemm_bf16_kernel<2><<<dim3(HID / 128, M / 128), 256, 0, stream>>>(
        ln2b, wfc1T, bfc1, nullptr, h1, M, HID, C);

    // FC2 + residual(xo) -> f32 output
    gemm_bf16_kernel<1><<<dim3(C / 128, M / 128), 256, 0, stream>>>(
        h1, wfc2T, bfc2, xo, out, M, C, HID);
}
